// KnowEvolveANHP_58703613002155
// MI455X (gfx1250) — matrix operand order repaired and checked by reading, not compiled, not run on hardware
//
#include <hip/hip_runtime.h>
#include <hip/hip_bf16.h>

#define BB 32
#define TT 64
#define BT 2048
#define DC 64
#define DL 256
#define DD 128
#define FEAT 320
#define K448 448
#define NH 4
#define DHd 80
#define NE 200000

typedef __attribute__((ext_vector_type(16))) _Float16 v16h;
typedef __attribute__((ext_vector_type(8)))  float    v8f;
typedef __attribute__((ext_vector_type(4)))  int      v4i;

__device__ __forceinline__ v8f wmma16(v16h a, v16h b, v8f c) {
  return __builtin_amdgcn_wmma_f32_16x16x32_f16(false, a, false, b, (short)0, c,
                                                false, false);
}

// A fragment (16x32 f16) from row-major [rows][ld]; lanes 0-15: row M=lane,
// element j holds K = 16*(lane>>4) + j.
__device__ __forceinline__ v16h load_a16(const _Float16* M, int ld, int row0,
                                         int k0, int lane) {
  int r = lane & 15;
  int kb = (lane >> 4) << 4;
  const _Float16* p = M + (size_t)(row0 + r) * ld + k0 + kb;
  v16h a;
#pragma unroll
  for (int j = 0; j < 8; ++j) { a[j] = p[j]; a[8 + j] = p[8 + j]; }
  return a;
}

__device__ __forceinline__ v16h load_a32(const float* M, int ld, int row0,
                                         int k0, int lane) {
  int r = lane & 15;
  int kb = (lane >> 4) << 4;
  const float* p = M + (size_t)(row0 + r) * ld + k0 + kb;
  v16h a;
#pragma unroll
  for (int j = 0; j < 8; ++j) {
    a[j] = (_Float16)p[j];
    a[8 + j] = (_Float16)p[8 + j];
  }
  return a;
}

// B fragment (32x16 f16) from row-major [K][ld]; lane holds col=lane&15,
// element j holds K = 16*(lane>>4) + j.
__device__ __forceinline__ v16h load_b16(const _Float16* M, int ld, int k0,
                                         int col0, int lane) {
  int c = lane & 15;
  int kb = (lane >> 4) << 4;
  const _Float16* p = M + (size_t)(k0 + kb) * ld + col0 + c;
  v16h b;
#pragma unroll
  for (int j = 0; j < 16; ++j) b[j] = p[(size_t)j * ld];
  return b;
}

// B = M^T for Q*K^T: rows of M (row-major [N][ld]) are B's columns -> contiguous j.
__device__ __forceinline__ v16h load_bT16(const _Float16* M, int ld, int col0,
                                          int k0, int lane) {
  int c = lane & 15;
  int kb = (lane >> 4) << 4;
  const _Float16* p = M + (size_t)(col0 + c) * ld + k0 + kb;
  v16h b;
#pragma unroll
  for (int j = 0; j < 16; ++j) b[j] = p[j];
  return b;
}

__device__ __forceinline__ float sigm(float x) { return 1.0f / (1.0f + __expf(-x)); }

// ---------------- utility kernels ----------------
__global__ void k_copy4(const float4* s, float4* d, int n) {
  int i = blockIdx.x * 256 + threadIdx.x;
  if (i < n) d[i] = s[i];
}
__global__ void k_copy1(const float* s, float* d, int n) {
  int i = blockIdx.x * 256 + threadIdx.x;
  if (i < n) d[i] = s[i];
}
__global__ void k_cvt(const float* s, _Float16* d, int n) {
  int i = blockIdx.x * 256 + threadIdx.x;
  if (i < n) d[i] = (_Float16)s[i];
}

// ---------------- phase 1: sequential scan (single workgroup) ----------------
__global__ void __launch_bounds__(256)
k_phase1(const int* ss, const int* so, const int* sr, const float* st,
         const float* rel, const float* Wst, const float* Wot,
         const _Float16* Whid, const _Float16* Whh,
         float* mem, float* lt, float* x) {
  __shared__ __attribute__((aligned(16))) _Float16 inp16[BB * FEAT];  // 20 KB
  __shared__ __attribute__((aligned(16))) _Float16 h16[BB * DL];      // 16 KB
  __shared__ float    hh[BB * DD];       // 16 KB
  __shared__ int   sidx[BB], oidx[BB], ridx[BB];
  __shared__ float tc[BB], dts[BB], dto[BB];

  int tid = threadIdx.x;
  int wave = tid >> 5, lane = tid & 31;

  for (int t = 0; t < TT; ++t) {
    if (tid < BB) {
      int b = tid;
      int s = ss[b * TT + t], o = so[b * TT + t], r = sr[b * TT + t];
      float tv = st[b * TT + t];
      sidx[b] = s; oidx[b] = o; ridx[b] = r; tc[b] = tv;
      dts[b] = tv - lt[s];
      dto[b] = tv - lt[o];
    }
    __syncthreads();
    // gather inp = [mem[s], mem[o], rel[r]]  (32 x 320) -> f16 LDS
    for (int i = tid; i < BB * FEAT; i += 256) {
      int b = i / FEAT, c = i - b * FEAT;
      float v;
      if (c < DD)            v = mem[(size_t)sidx[b] * DD + c];
      else if (c < 2 * DD)   v = mem[(size_t)oidx[b] * DD + (c - DD)];
      else                   v = rel[(size_t)ridx[b] * DC + (c - 2 * DD)];
      inp16[i] = (_Float16)v;
    }
    __syncthreads();
    // h = sigmoid(inp @ W_hidden): M=32 N=256 K=320 -> 32 tiles / 8 waves
    for (int tt2 = wave; tt2 < 32; tt2 += 8) {
      int mt = tt2 >> 4, nt = tt2 & 15;
      v8f acc = {};
      for (int kt = 0; kt < 10; ++kt) {
        v16h a = load_a16(inp16, FEAT, mt * 16, kt * 32, lane);
        v16h b = load_b16(Whid, DL, kt * 32, nt * 16, lane);
        acc = wmma16(a, b, acc);
      }
      int col = nt * 16 + (lane & 15);
      int rbase = mt * 16 + (lane >> 4) * 8;
#pragma unroll
      for (int i = 0; i < 8; ++i)
        h16[(rbase + i) * DL + col] = (_Float16)sigm(acc[i]);
    }
    __syncthreads();
    // hh = h @ W_hh: M=32 N=128 K=256 -> 16 tiles / 8 waves
    for (int tt2 = wave; tt2 < 16; tt2 += 8) {
      int mt = tt2 >> 3, nt = tt2 & 7;
      v8f acc = {};
      for (int kt = 0; kt < 8; ++kt) {
        v16h a = load_a16(h16, DL, mt * 16, kt * 32, lane);
        v16h b = load_b16(Whh, DD, kt * 32, nt * 16, lane);
        acc = wmma16(a, b, acc);
      }
      int col = nt * 16 + (lane & 15);
      int rbase = mt * 16 + (lane >> 4) * 8;
#pragma unroll
      for (int i = 0; i < 8; ++i) hh[(rbase + i) * DD + col] = acc[i];
    }
    __syncthreads();
    // elementwise gate + subject scatter + evt write; stash new_o in hh
    for (int i = tid; i < BB * DD; i += 256) {
      int b = i >> 7, d = i & 127;
      float hv = hh[i];
      float ns = sigm(dts[b] * Wst[d] + hv);
      float no = sigm(dto[b] * Wot[d] + hv);
      mem[(size_t)sidx[b] * DD + d] = ns;  // subject first
      size_t xr = (size_t)(b * TT + t) * FEAT;
      x[xr + d] = ns;
      x[xr + DD + d] = no;
      hh[i] = no;
    }
    for (int i = tid; i < BB * DC; i += 256) {
      int b = i >> 6, c = i & 63;
      x[(size_t)(b * TT + t) * FEAT + 2 * DD + c] = rel[(size_t)ridx[b] * DC + c];
    }
    if (tid < BB) lt[sidx[tid]] = tc[tid];
    __threadfence();
    __syncthreads();
    // object scatter second (object wins s==o ties, matching .at[s].set().at[o].set())
    for (int i = tid; i < BB * DD; i += 256) {
      int b = i >> 7, d = i & 127;
      mem[(size_t)oidx[b] * DD + d] = hh[i];
    }
    if (tid < BB) lt[oidx[tid]] = tc[tid];
    __threadfence();
    __syncthreads();
  }
}

// ---------------- build e = [x, time_emb] as f16 (ones handled as bias row 448) ----
__global__ void k_build_e(const float* x, const float* st, _Float16* e) {
  int idx = blockIdx.x * 256 + threadIdx.x;
  if (idx >= BT * K448) return;
  int row = idx / K448, c = idx - row * K448;
  float val;
  if (c < FEAT) {
    val = x[(size_t)row * FEAT + c];
  } else {
    int d = c - FEAT;
    int b = row >> 6, t = row & 63;
    float tm = st[b * TT + t];
    val = __sinf(tm * __powf(1.0e4f, -(float)d));  // norm = (5*TIME_MAX/TIME_M)^-d
  }
  e[idx] = (_Float16)val;
}

// ---------------- QKV projections: [2048,448] x [448,320] + bias(row 448) --------
__global__ void __launch_bounds__(256)
k_qkv(const _Float16* e, const _Float16* Wq, const _Float16* Wk,
      const _Float16* Wv, _Float16* q, _Float16* k, _Float16* v) {
  int wave = threadIdx.x >> 5, lane = threadIdx.x & 31;
  const _Float16* W;
  _Float16* out;
  if (blockIdx.z == 0)      { W = Wq; out = q; }
  else if (blockIdx.z == 1) { W = Wk; out = k; }
  else                      { W = Wv; out = v; }
  int rb = blockIdx.x * 64, cb = blockIdx.y * 64;
  for (int tt = wave; tt < 16; tt += 8) {
    int row0 = rb + (tt >> 2) * 16, col0 = cb + (tt & 3) * 16;
    v8f acc = {};
    for (int kt = 0; kt < 14; ++kt) {
      // keep the next weight K-slab warm in GL2 (global_prefetch_b8)
      if (kt < 13)
        __builtin_prefetch(W + (size_t)(kt + 1) * 32 * FEAT + col0, 0, 1);
      v16h a = load_a16(e, K448, row0, kt * 32, lane);
      v16h b = load_b16(W, FEAT, kt * 32, col0, lane);
      acc = wmma16(a, b, acc);
    }
    int col = col0 + (lane & 15);
    float bias = (float)W[(size_t)K448 * FEAT + col];  // ones-column contribution
    int rbase = row0 + (lane >> 4) * 8;
#pragma unroll
    for (int i = 0; i < 8; ++i)
      out[(size_t)(rbase + i) * FEAT + col] = (_Float16)(acc[i] + bias);
  }
}

// ---------------- attention core per (b,h): QK^T, causal softmax, AV ------------
__global__ void __launch_bounds__(256)
k_attn(const _Float16* q, const _Float16* k, const _Float16* v, _Float16* ao) {
  __shared__ __attribute__((aligned(16))) _Float16 qs[64 * 96];  // DH=80 pad 96
  __shared__ __attribute__((aligned(16))) _Float16 ks[64 * 96];
  __shared__ float    sc[64 * 64];
  __shared__ __attribute__((aligned(16))) _Float16 at[64 * 64];
  int tid = threadIdx.x, wave = tid >> 5, lane = tid & 31;
  int b = blockIdx.x >> 2, h = blockIdx.x & 3;
  size_t base = (size_t)b * 64 * FEAT + (size_t)h * DHd;

#if __has_builtin(__builtin_amdgcn_global_load_async_to_lds_b128)
  // zero the K-padding columns (80..95)
  for (int i = tid; i < 64 * 96; i += 256) {
    if ((i % 96) >= DHd) { qs[i] = (_Float16)0.0f; ks[i] = (_Float16)0.0f; }
  }
  // direct L2 -> LDS async copy, 16B per op: 64 rows x 10 segs x {q,k}
  for (int i = tid; i < 64 * 10 * 2; i += 256) {
    int which = (i >= 640) ? 1 : 0;
    int j = i - which * 640;
    int r = j / 10, seg = j - r * 10;
    const _Float16* g = (which ? k : q) + base + (size_t)r * FEAT + seg * 8;
    _Float16* l = (which ? ks : qs) + r * 96 + seg * 8;
    __builtin_amdgcn_global_load_async_to_lds_b128(
        (__attribute__((address_space(1))) v4i*)(void*)g,
        (__attribute__((address_space(3))) v4i*)(void*)l, 0, 0);
  }
  asm volatile("s_wait_asynccnt 0" ::: "memory");
  __syncthreads();
#else
  for (int i = tid; i < 64 * 96; i += 256) {
    int r = i / 96, c = i - r * 96;
    _Float16 qv = (_Float16)0.0f, kv = (_Float16)0.0f;
    if (c < DHd) {
      qv = q[base + (size_t)r * FEAT + c];
      kv = k[base + (size_t)r * FEAT + c];
    }
    qs[i] = qv;
    ks[i] = kv;
  }
  __syncthreads();
#endif

  const float scale = 0.11180339887498948f;  // 1/sqrt(80)
  for (int tt = wave; tt < 16; tt += 8) {
    int mt = tt >> 2, nt = tt & 3;
    v8f acc = {};
    for (int kt = 0; kt < 3; ++kt) {
      v16h a = load_a16(qs, 96, mt * 16, kt * 32, lane);
      v16h bb = load_bT16(ks, 96, nt * 16, kt * 32, lane);
      acc = wmma16(a, bb, acc);
    }
    int col = nt * 16 + (lane & 15);
    int rbase = mt * 16 + (lane >> 4) * 8;
#pragma unroll
    for (int i = 0; i < 8; ++i) sc[(rbase + i) * 64 + col] = acc[i] * scale;
  }
  __syncthreads();
  if (tid < 64) {  // strict-causal softmax; row 0 -> all zeros
    int r = tid;
    float m = -3.0e38f;
    for (int j = 0; j < r; ++j) m = fmaxf(m, sc[r * 64 + j]);
    float ssum = 0.0f;
    for (int j = 0; j < r; ++j) ssum += __expf(sc[r * 64 + j] - m);
    float inv = (r > 0) ? 1.0f / ssum : 0.0f;
    for (int j = 0; j < 64; ++j)
      at[r * 64 + j] =
          (_Float16)((j < r) ? __expf(sc[r * 64 + j] - m) * inv : 0.0f);
  }
  __syncthreads();
  for (int tt = wave; tt < 20; tt += 8) {  // [64,64] x [64,80]
    int mt = tt / 5, nt = tt - mt * 5;
    v8f acc = {};
    for (int kt = 0; kt < 2; ++kt) {
      v16h a = load_a16(at, 64, mt * 16, kt * 32, lane);
      v16h bb = load_b16(v + base, FEAT, kt * 32, nt * 16, lane);
      acc = wmma16(a, bb, acc);
    }
    int col = nt * 16 + (lane & 15);
    int rbase = mt * 16 + (lane >> 4) * 8;
#pragma unroll
    for (int i = 0; i < 8; ++i)
      ao[base + (size_t)(rbase + i) * FEAT + col] = (_Float16)acc[i];
  }
}

// ---------------- x += tanh(ao @ Wo) ----------------
__global__ void __launch_bounds__(256)
k_wo(const _Float16* ao, const _Float16* W, float* x) {
  int wave = threadIdx.x >> 5, lane = threadIdx.x & 31;
  int rb = blockIdx.x * 64, cb = blockIdx.y * 64;
  for (int tt = wave; tt < 16; tt += 8) {
    int row0 = rb + (tt >> 2) * 16, col0 = cb + (tt & 3) * 16;
    v8f acc = {};
    for (int kt = 0; kt < 10; ++kt) {
      if (kt < 9)
        __builtin_prefetch(W + (size_t)(kt + 1) * 32 * FEAT + col0, 0, 1);
      v16h a = load_a16(ao, FEAT, row0, kt * 32, lane);
      v16h b = load_b16(W, FEAT, kt * 32, col0, lane);
      acc = wmma16(a, b, acc);
    }
    int col = col0 + (lane & 15);
    int rbase = row0 + (lane >> 4) * 8;
#pragma unroll
    for (int i = 0; i < 8; ++i) {
      size_t idx = (size_t)(rbase + i) * FEAT + col;
      x[idx] += tanhf(acc[i]);
    }
  }
}

// ---------------- score MLP + softplus + partial loss ----------------
__global__ void __launch_bounds__(256)
k_score(const float* x, const _Float16* W1, const float* b1, const float* W2,
        const float* b2, float* lam, float* part) {
  __shared__ float hid[64 * DD];  // 32 KB
  __shared__ float red[64];
  int tid = threadIdx.x, wave = tid >> 5, lane = tid & 31;
  int rb = blockIdx.x * 64;
  for (int tt = wave; tt < 32; tt += 8) {
    int mt = tt >> 3, nt = tt & 7;
    v8f acc = {};
    for (int kt = 0; kt < 10; ++kt) {
      v16h a = load_a32(x, FEAT, rb + mt * 16, kt * 32, lane);
      v16h b = load_b16(W1, DD, kt * 32, nt * 16, lane);
      acc = wmma16(a, b, acc);
    }
    int col = nt * 16 + (lane & 15);
    int rbase = mt * 16 + (lane >> 4) * 8;
    float bias = b1[col];
#pragma unroll
    for (int i = 0; i < 8; ++i)
      hid[(rbase + i) * DD + col] = fmaxf(acc[i] + bias, 0.0f);
  }
  __syncthreads();
  if (tid < 64) {
    float s = b2[0];
    for (int d = 0; d < DD; ++d) s += hid[tid * DD + d] * W2[d];
    float lm = (s > 20.0f) ? s : log1pf(__expf(s));
    lam[rb + tid] = lm;
    red[tid] = -logf(lm + 1e-8f) * (1.0f / (float)BT);
  }
  __syncthreads();
  if (tid == 0) {
    float s = 0.0f;
    for (int i = 0; i < 64; ++i) s += red[i];
    part[blockIdx.x] = s;
  }
}

__global__ void k_loss_sum(const float* part, float* out_loss) {
  if (threadIdx.x == 0) {
    float s = 0.0f;
    for (int i = 0; i < 32; ++i) s += part[i];
    out_loss[0] = s;
  }
}

// ---------------- host launcher ----------------
extern "C" void kernel_launch(void* const* d_in, const int* in_sizes, int n_in,
                              void* d_out, int out_size, void* d_ws,
                              size_t ws_size, hipStream_t stream) {
  (void)in_sizes; (void)n_in; (void)out_size; (void)ws_size;
  const int*   seq_s  = (const int*)d_in[0];
  const int*   seq_o  = (const int*)d_in[1];
  const int*   seq_r  = (const int*)d_in[2];
  const float* seq_t  = (const float*)d_in[3];
  const float* mem_in = (const float*)d_in[4];
  const float* lt_in  = (const float*)d_in[5];
  const float* rel    = (const float*)d_in[6];
  const float* Whid   = (const float*)d_in[7];
  const float* Whh    = (const float*)d_in[8];
  const float* Wst    = (const float*)d_in[9];
  const float* Wot    = (const float*)d_in[10];
  const float* Wq     = (const float*)d_in[11];
  const float* Wk     = (const float*)d_in[12];
  const float* Wv     = (const float*)d_in[13];
  const float* Wo     = (const float*)d_in[14];
  const float* sW1    = (const float*)d_in[15];
  const float* sb1    = (const float*)d_in[16];
  const float* sW2    = (const float*)d_in[17];
  const float* sb2    = (const float*)d_in[18];

  float* out_loss = (float*)d_out;
  float* out_lam  = out_loss + 1;
  float* out_mem  = out_lam + BT;
  float* out_lt   = out_mem + (size_t)NE * DD;

  char* w = (char*)d_ws;
  size_t off = 0;
  auto alloc = [&](size_t bytes) {
    void* p = w + off;
    off = (off + bytes + 255) & ~(size_t)255;
    return p;
  };
  float*    x      = (float*)alloc((size_t)BT * FEAT * 4);
  _Float16* e16    = (_Float16*)alloc((size_t)BT * K448 * 2);
  _Float16* q16    = (_Float16*)alloc((size_t)BT * FEAT * 2);
  _Float16* k16    = (_Float16*)alloc((size_t)BT * FEAT * 2);
  _Float16* v16p   = (_Float16*)alloc((size_t)BT * FEAT * 2);
  _Float16* ao16   = (_Float16*)alloc((size_t)BT * FEAT * 2);
  _Float16* Whid16 = (_Float16*)alloc((size_t)FEAT * DL * 2);
  _Float16* Whh16  = (_Float16*)alloc((size_t)DL * DD * 2);
  _Float16* Wq16   = (_Float16*)alloc((size_t)2 * 449 * FEAT * 2);
  _Float16* Wk16   = (_Float16*)alloc((size_t)2 * 449 * FEAT * 2);
  _Float16* Wv16   = (_Float16*)alloc((size_t)2 * 449 * FEAT * 2);
  _Float16* Wo16   = (_Float16*)alloc((size_t)2 * FEAT * FEAT * 2);
  _Float16* W1_16  = (_Float16*)alloc((size_t)FEAT * DD * 2);
  float*    part   = (float*)alloc(32 * 4);

  // bulk copy of persistent buffers into output (dominant HBM traffic)
  int n4 = NE * DD / 4;
  k_copy4<<<(n4 + 255) / 256, 256, 0, stream>>>((const float4*)mem_in,
                                                (float4*)out_mem, n4);
  k_copy1<<<(NE + 255) / 256, 256, 0, stream>>>(lt_in, out_lt, NE);

  // one-time f32 -> f16 weight conversion
  k_cvt<<<(FEAT * DL + 255) / 256, 256, 0, stream>>>(Whid, Whid16, FEAT * DL);
  k_cvt<<<(DL * DD + 255) / 256, 256, 0, stream>>>(Whh, Whh16, DL * DD);
  int nqkv = 2 * 449 * FEAT;
  k_cvt<<<(nqkv + 255) / 256, 256, 0, stream>>>(Wq, Wq16, nqkv);
  k_cvt<<<(nqkv + 255) / 256, 256, 0, stream>>>(Wk, Wk16, nqkv);
  k_cvt<<<(nqkv + 255) / 256, 256, 0, stream>>>(Wv, Wv16, nqkv);
  int nwo = 2 * FEAT * FEAT;
  k_cvt<<<(nwo + 255) / 256, 256, 0, stream>>>(Wo, Wo16, nwo);
  k_cvt<<<(FEAT * DD + 255) / 256, 256, 0, stream>>>(sW1, W1_16, FEAT * DD);

  // sequential RNN scan over T (single persistent workgroup, WMMA per step)
  k_phase1<<<1, 256, 0, stream>>>(seq_s, seq_o, seq_r, seq_t, rel, Wst, Wot,
                                  Whid16, Whh16, out_mem, out_lt, x);

  // two attention layers
  for (int l = 0; l < 2; ++l) {
    k_build_e<<<(BT * K448 + 255) / 256, 256, 0, stream>>>(x, seq_t, e16);
    k_qkv<<<dim3(32, 5, 3), 256, 0, stream>>>(
        e16, Wq16 + (size_t)l * 449 * FEAT, Wk16 + (size_t)l * 449 * FEAT,
        Wv16 + (size_t)l * 449 * FEAT, q16, k16, v16p);
    k_attn<<<BB * NH, 256, 0, stream>>>(q16, k16, v16p, ao16);
    k_wo<<<dim3(32, 5), 256, 0, stream>>>(ao16, Wo16 + (size_t)l * FEAT * FEAT,
                                          x);
  }

  // score MLP -> lam, deterministic two-stage loss reduction
  k_score<<<32, 256, 0, stream>>>(x, W1_16, sb1, sW2, sb2, out_lam, part);
  k_loss_sum<<<1, 32, 0, stream>>>(part, out_loss);
}
